// EnhancedPathReconstructor_77352361001299
// MI455X (gfx1250) — matrix operand order repaired and checked by reading, not compiled, not run on hardware
//
#include <hip/hip_runtime.h>
#include <hip/hip_fp16.h>
#include <cstdint>

typedef __attribute__((ext_vector_type(16))) _Float16 v16h;
typedef __attribute__((ext_vector_type(8)))  float    v8f;

#define BB 8
#define NN 2048
#define HH 128

// ---------------------------------------------------------------------------
// Kernel 1: A = emb @ W1[:H],  C = emb @ W1[H:] + b1   via v_wmma_f32_16x16x32_f16
// grid = (1024 M-tiles, 8 N-tiles, 2 outputs), block = 32 (one wave per tile)
// ---------------------------------------------------------------------------
__global__ void epr_gemm_wmma(const float* __restrict__ emb,
                              const float* __restrict__ W1,
                              const float* __restrict__ b1,
                              float* __restrict__ outA,
                              float* __restrict__ outC) {
  const int lane = threadIdx.x & 31;
  const int half = lane >> 4;      // which half-wave
  const int l16  = lane & 15;
  const int mbase = blockIdx.x * 16;
  const int ncol  = blockIdx.y * 16 + l16;
  const int which = blockIdx.z;    // 0 -> W1a, 1 -> W1b(+b1)
  const float* W = W1 + (size_t)which * HH * HH;       // [128][128] slice, input-major
  const float* arow = emb + (size_t)(mbase + l16) * HH;

  v8f acc = {};
  #pragma unroll
  for (int kc = 0; kc < 4; ++kc) {                     // K = 128 in chunks of 32
    const int k0 = kc * 32;
    v16h afrag, bfrag;
    // A-frag (16x32 f16): lane holds row M=l16; element (v,lo/hi):
    //   K = (v>>2)*16 + half*8 + (v&3)*2 + (e&1)
    #pragma unroll
    for (int v = 0; v < 8; ++v) {
      const int K = k0 + ((v >> 2) << 4) + (half << 3) + ((v & 3) << 1);
      const float2 av = *(const float2*)(arow + K);
      afrag[2 * v]     = (_Float16)av.x;
      afrag[2 * v + 1] = (_Float16)av.y;
    }
    // B-frag (32x16 f16): lane holds col N=l16; elements index K = (e>>3)*16 + half*8 + (e&7)
    #pragma unroll
    for (int e = 0; e < 16; ++e) {
      const int K = k0 + ((e >> 3) << 4) + (half << 3) + (e & 7);
      bfrag[e] = (_Float16)W[(size_t)K * HH + ncol];
    }
    acc = __builtin_amdgcn_wmma_f32_16x16x32_f16(false, afrag, false, bfrag,
                                                 (short)0, acc, false, false);
  }

  const float bias = which ? b1[ncol] : 0.0f;
  float* outp = which ? outC : outA;
  // C/D layout: VGPR i holds (M = i + half*8, N = l16)
  #pragma unroll
  for (int i = 0; i < 8; ++i) {
    const int M = mbase + (half << 3) + i;
    outp[(size_t)M * HH + ncol] = acc[i] + bias;
  }
}

// ---------------------------------------------------------------------------
// Kernel 2: full pairwise logit matrix
//   S[b,i,j] = sum_h w2[h] * elu(A[b,i,h] + C[b,j,h]) + b2
// One block per (b,i) row; 256 threads sweep j. VALU/exp bound.
// ---------------------------------------------------------------------------
__global__ void epr_pair_scores(const float* __restrict__ A,
                                const float* __restrict__ C,
                                const float* __restrict__ w2,
                                const float* __restrict__ b2p,
                                float* __restrict__ S) {
  __shared__ float a_sh[HH];
  __shared__ float w_sh[HH];
  const int bi  = blockIdx.x;       // b*N + i
  const int b   = bi >> 11;
  const int tid = threadIdx.x;
  if (tid < HH)          a_sh[tid]      = A[(size_t)bi * HH + tid];
  else if (tid < 2 * HH) w_sh[tid - HH] = w2[tid - HH];
  __syncthreads();

  const float  b2v = b2p[0];
  const float* Cb  = C + (size_t)b * NN * HH;
  float*       Srw = S + (size_t)bi * NN;

  for (int j = tid; j < NN; j += 256) {
    const float4* cp = (const float4*)(Cb + (size_t)j * HH);
    float sum = 0.0f;
    #pragma unroll 8
    for (int q = 0; q < HH / 4; ++q) {
      const float4 c = cp[q];
      float x0 = a_sh[4 * q + 0] + c.x;
      float x1 = a_sh[4 * q + 1] + c.y;
      float x2 = a_sh[4 * q + 2] + c.z;
      float x3 = a_sh[4 * q + 3] + c.w;
      x0 = (x0 > 0.0f) ? x0 : (__expf(x0) - 1.0f);
      x1 = (x1 > 0.0f) ? x1 : (__expf(x1) - 1.0f);
      x2 = (x2 > 0.0f) ? x2 : (__expf(x2) - 1.0f);
      x3 = (x3 > 0.0f) ? x3 : (__expf(x3) - 1.0f);
      sum = fmaf(w_sh[4 * q + 0], x0, sum);
      sum = fmaf(w_sh[4 * q + 1], x1, sum);
      sum = fmaf(w_sh[4 * q + 2], x2, sum);
      sum = fmaf(w_sh[4 * q + 3], x3, sum);
    }
    Srw[j] = sum + b2v;
  }
}

// ---------------------------------------------------------------------------
// Kernel 3: sequential greedy scan. One block (32 waves) per batch element.
// Per step: masked argmax over S[b,cur,:] (L2-resident 8KB row), 2 barriers.
// ---------------------------------------------------------------------------
__global__ __launch_bounds__(1024) void epr_scan(const float* __restrict__ emb,
                                                 const float* __restrict__ S,
                                                 float* __restrict__ out_path,
                                                 float* __restrict__ out_scores) {
  const int b    = blockIdx.x;
  const int tid  = threadIdx.x;
  const int lane = tid & 31;
  const int wave = tid >> 5;
  const float NEG = -__builtin_inff();

  __shared__ unsigned vis[NN / 32];
  __shared__ float rs[32];
  __shared__ int   ri[32];
  __shared__ int   s_cur;
  __shared__ int   s_active;

  if (tid < NN / 32) vis[tid] = 0u;

  // ---- root = argmax_n emb[b,n,0] (first max) ----
  float v0 = emb[(size_t)(b * NN + tid) * HH];
  float v1 = emb[(size_t)(b * NN + tid + 1024) * HH];
  float bv = v0;
  int   bx = tid;
  if (v1 > bv) { bv = v1; bx = tid + 1024; }
  #pragma unroll
  for (int off = 16; off > 0; off >>= 1) {
    float ov = __shfl_xor(bv, off, 32);
    int   oi = __shfl_xor(bx, off, 32);
    if (ov > bv || (ov == bv && oi < bx)) { bv = ov; bx = oi; }
  }
  if (lane == 0) { rs[wave] = bv; ri[wave] = bx; }
  __syncthreads();
  if (tid < 32) {
    bv = rs[tid]; bx = ri[tid];
    #pragma unroll
    for (int off = 16; off > 0; off >>= 1) {
      float ov = __shfl_xor(bv, off, 32);
      int   oi = __shfl_xor(bx, off, 32);
      if (ov > bv || (ov == bv && oi < bx)) { bv = ov; bx = oi; }
    }
    if (tid == 0) {
      s_cur    = bx;
      s_active = 1;
      vis[bx >> 5] |= (1u << (bx & 31));
      out_path[(size_t)b * NN]   = (float)bx;   // path as numeric cast
      out_scores[(size_t)b * NN] = 1.0f;        // root score = 1.0
    }
  }

  // ---- 2047 greedy steps ----
  for (int step = 1; step < NN; ++step) {
    __syncthreads();
    if (!s_active) {
      for (int s2 = step + tid; s2 < NN; s2 += 1024) {
        out_path[(size_t)b * NN + s2]   = -1.0f;
        out_scores[(size_t)b * NN + s2] = 0.0f;
      }
      break;
    }
    const int cur = s_cur;
    const float* Srow = S + ((size_t)b * NN + cur) * NN;

    float c0 = Srow[tid];
    float c1 = Srow[tid + 1024];
    if ((vis[tid >> 5] >> (tid & 31)) & 1u) c0 = NEG;
    const int n1 = tid + 1024;
    if ((vis[n1 >> 5] >> (n1 & 31)) & 1u) c1 = NEG;

    bv = c0; bx = tid;
    if (c1 > bv) { bv = c1; bx = n1; }
    #pragma unroll
    for (int off = 16; off > 0; off >>= 1) {
      float ov = __shfl_xor(bv, off, 32);
      int   oi = __shfl_xor(bx, off, 32);
      if (ov > bv || (ov == bv && oi < bx)) { bv = ov; bx = oi; }
    }
    if (lane == 0) { rs[wave] = bv; ri[wave] = bx; }
    __syncthreads();
    if (tid < 32) {
      bv = rs[tid]; bx = ri[tid];
      #pragma unroll
      for (int off = 16; off > 0; off >>= 1) {
        float ov = __shfl_xor(bv, off, 32);
        int   oi = __shfl_xor(bx, off, 32);
        if (ov > bv || (ov == bv && oi < bx)) { bv = ov; bx = oi; }
      }
      if (tid == 0) {
        const float sc = 1.0f / (1.0f + __expf(-bv));   // sigmoid(logit)
        if (sc > 0.3f) {                                 // THRESH
          vis[bx >> 5] |= (1u << (bx & 31));
          s_cur = bx;
          out_path[(size_t)b * NN + step]   = (float)bx;
          out_scores[(size_t)b * NN + step] = sc;
        } else {
          s_active = 0;
          out_path[(size_t)b * NN + step]   = -1.0f;
          out_scores[(size_t)b * NN + step] = 0.0f;
        }
      }
    }
  }
}

// ---------------------------------------------------------------------------
extern "C" void kernel_launch(void* const* d_in, const int* in_sizes, int n_in,
                              void* d_out, int out_size, void* d_ws, size_t ws_size,
                              hipStream_t stream) {
  (void)in_sizes; (void)n_in; (void)out_size; (void)ws_size;
  const float* emb = (const float*)d_in[0];
  // d_in[1] = batch (int64), unused: graphs are contiguous & equal-sized
  const float* W1  = (const float*)d_in[2];   // [256,128] input-major
  const float* b1  = (const float*)d_in[3];   // [128]
  const float* w2  = (const float*)d_in[4];   // [128]
  const float* b2  = (const float*)d_in[5];   // scalar

  float* out = (float*)d_out;                 // path [B,N] then scores [B,N]
  float* ws  = (float*)d_ws;
  float* A = ws;                                        // 8  MB
  float* C = A + (size_t)BB * NN * HH;                  // 8  MB
  float* S = C + (size_t)BB * NN * HH;                  // 128 MB (fits L2)

  hipLaunchKernelGGL(epr_gemm_wmma, dim3(BB * NN / 16, HH / 16, 2), dim3(32),
                     0, stream, emb, W1, b1, A, C);
  hipLaunchKernelGGL(epr_pair_scores, dim3(BB * NN), dim3(256),
                     0, stream, A, C, w2, b2, S);
  hipLaunchKernelGGL(epr_scan, dim3(BB), dim3(1024),
                     0, stream, emb, S, out, out + (size_t)BB * NN);
}
